// SphericalFourierNeuralOperatorNet_84997402788256
// MI455X (gfx1250) — matrix operand order repaired and checked by reading, not compiled, not run on hardware
//
#include <hip/hip_runtime.h>
#include <hip/hip_bf16.h>
#include <stdint.h>

// ---------------------------------------------------------------------------
// SFNO forward on MI455X (gfx1250, wave32).
// Every contraction is a f16->f32 WMMA GEMM (v_wmma_f32_16x16x32_f16).
// rfft/irfft are dense cos/sin f16 table GEMMs; Legendre + spectral filters
// are batched GEMMs. All operand layouts are inner-stride-1, so the GEMM has
// no strided fallback paths at all. Double-buffered LDS pipeline: tile k+1's
// A streams via global_load_async_to_lds_b128 (ASYNCcnt) and B global loads
// sit in registers while tile k's WMMAs run; one barrier per K-step.
// Complex products use a two-term GEMM (minus sign = 0x8000 XOR on B2).
// Block tile 128(M) x 64(N) x 32(K), 8 waves, wave tile 32x32 (4 wmma accs).
// Workspace: ~370 MB f16 scratch.
// ---------------------------------------------------------------------------

typedef _Float16 h16;
typedef __attribute__((ext_vector_type(16))) _Float16 v16h;
typedef __attribute__((ext_vector_type(8)))  float    v8f;

#if defined(__has_builtin)
#  if __has_builtin(__builtin_amdgcn_global_load_async_to_lds_b128)
#    define ASYNC_LDS 1
#  endif
#  if __has_builtin(__builtin_amdgcn_s_wait_asynccnt)
#    define WAIT_ASYNC() __builtin_amdgcn_s_wait_asynccnt(0)
#  else
#    define WAIT_ASYNC() asm volatile("s_wait_asynccnt 0" ::: "memory")
#  endif
#endif
#ifndef ASYNC_LDS
#  define ASYNC_LDS 0
#endif
#ifndef WAIT_ASYNC
#  define WAIT_ASYNC() do {} while (0)
#endif

#if ASYNC_LDS
// Builtin signature (from clang diagnostic): param0 = v4i addrspace(1)* (global,
// non-const), param1 = v4i addrspace(3)* (LDS), then imm offset + cpol.
typedef int v4i __attribute__((vector_size(16)));
typedef __attribute__((address_space(1))) v4i* as1_v4i;
typedef __attribute__((address_space(3))) v4i* as3_v4i;
__device__ __forceinline__ void async_cp16(const h16* g, h16* l) {
  v4i* pg = (v4i*)(uintptr_t)g;   // strip const via integer round-trip
  v4i* pl = (v4i*)(uintptr_t)l;
  __builtin_amdgcn_global_load_async_to_lds_b128((as1_v4i)pg, (as3_v4i)pl, 0, 0);
}
#endif

#define WMMA16(a, b, c) \
  __builtin_amdgcn_wmma_f32_16x16x32_f16(false, (a), false, (b), (short)0, (c), false, false)

__device__ __forceinline__ float gelu_f(float v) {
  return 0.5f * v * (1.0f + erff(v * 0.70710678118654752f));
}

// A fragment (16x32 f16): lane = M row (lane&15), K halves {kb..kb+7, kb+16..kb+23},
// kb = 0 for lanes 0-15, 8 for lanes 16-31.  LDS tile is [128 rows][32 k].
__device__ __forceinline__ v16h frag_a(const h16* s, int row0, int lane) {
  int m  = row0 + (lane & 15);
  int kb = (lane < 16) ? 0 : 8;
  const h16* p = s + m * 32 + kb;
  v16h r;
  ((uint4*)&r)[0] = *(const uint4*)(p);
  ((uint4*)&r)[1] = *(const uint4*)(p + 16);
  return r;
}

// B fragment (32x16 f16): lane = N col (lane&15), K halves {kb..kb+7, kb+16..kb+23}, kb = (lane<16?0:8).
// LDS tile stored transposed [64 n][32 k] -> each lane reads two runs of 8 contiguous halfs.
__device__ __forceinline__ v16h frag_b(const h16* s, int col0, int lane) {
  int n  = col0 + (lane & 15);
  int kb = (lane < 16) ? 0 : 8;
  const h16* p = s + n * 32 + kb;
  v16h r;
  ((uint4*)&r)[0] = *(const uint4*)(p);
  ((uint4*)&r)[1] = *(const uint4*)(p + 16);
  return r;
}

// ---------------------------------------------------------------------------
// Workhorse GEMM:  D[M,N] (f16 out, f32 acc) = A1*B1 (+ A2*(+/-B2))
// A[m,k] at A + offA*z + m*sAm + k   (inner K stride must be 1)
// B[k,n] at B + offB*z + k*sBk + n   (inner N stride must be 1)
// M multiple of 128; K multiple of 32; stores guarded by n < Nreal (last N
// tile may read into workspace slack). epi: 0=none 1=gelu 2=+f16aux 3=+f32aux.
// ---------------------------------------------------------------------------
template<bool TWO>
__global__ __launch_bounds__(256) void gemm_f16(
    const h16* __restrict__ A1, long offA, long sAm,
    const h16* __restrict__ B1, long offB, long sBk,
    const h16* __restrict__ A2, const h16* __restrict__ B2, unsigned negMask,
    h16* __restrict__ D, long offD, long sDm, long sDn,
    int Nreal, int K,
    int epi, const void* __restrict__ aux, long offX, long sXm, long sXn)
{
  __shared__ h16 sA1[2][128 * 32];
  __shared__ h16 sB1[2][64 * 32];
  __shared__ h16 sA2s[2][TWO ? 128 * 32 : 16];
  __shared__ h16 sB2s[2][TWO ? 64 * 32 : 16];

  const int tid  = threadIdx.x;
  const int lane = tid & 31;
  const int wave = tid >> 5;
  const int wm   = wave >> 1;   // 0..3 -> M sub-tile (32 rows)
  const int wn   = wave & 1;    // 0..1 -> N sub-tile (32 cols)
  const long z   = blockIdx.z;
  const int rowBase = blockIdx.y * 128;
  const int colBase = blockIdx.x * 64;

  // A staging: thread covers row r, halfs kh..kh+15 of the 128x32 tile.
  const int  r  = tid >> 1;
  const int  kh = (tid & 1) * 16;
  const long aOff = offA * z + (long)(rowBase + r) * sAm + kh;
  // B staging: thread covers k-row bk, cols bn..bn+7 of the 32x64 tile.
  const int  bk = tid >> 3;
  const int  bn = (tid & 7) * 8;
  const long bOff = offB * z + (long)bk * sBk + colBase + bn;

  v8f acc[2][2];
  for (int i = 0; i < 2; ++i)
    for (int j = 0; j < 2; ++j)
      acc[i][j] = (v8f){0.f, 0.f, 0.f, 0.f, 0.f, 0.f, 0.f, 0.f};

  uint4 rA0 = {}, rA1 = {}, rA20 = {}, rA21 = {}, rB = {}, rB2 = {};

  // issue global loads for tile kt (A goes straight to LDS via async DMA)
  auto stage_issue = [&](int kt, int buf) {
    long ga = aOff + (long)kt * 32;
#if ASYNC_LDS
    h16* dA = sA1[buf] + r * 32 + kh;
    async_cp16(A1 + ga, dA);
    async_cp16(A1 + ga + 8, dA + 8);
#else
    rA0 = ((const uint4*)(A1 + ga))[0];
    rA1 = ((const uint4*)(A1 + ga))[1];
#endif
    __builtin_prefetch(A1 + ga + 64, 0, 1);      // global_prefetch_b8
    if (TWO) {
#if ASYNC_LDS
      h16* d2 = sA2s[buf] + r * 32 + kh;
      async_cp16(A2 + ga, d2);
      async_cp16(A2 + ga + 8, d2 + 8);
#else
      rA20 = ((const uint4*)(A2 + ga))[0];
      rA21 = ((const uint4*)(A2 + ga))[1];
#endif
    }
    long gb = bOff + (long)kt * 32 * sBk;
    rB = *(const uint4*)(B1 + gb);
    __builtin_prefetch(B1 + gb + (long)32 * sBk, 0, 1);
    if (TWO) rB2 = *(const uint4*)(B2 + gb);
    (void)buf;
  };

  // complete staging: transpose-store B (sign of term 2 applied here)
  auto stage_store = [&](int buf) {
#if !ASYNC_LDS
    h16* dA = sA1[buf] + r * 32 + kh;
    ((uint4*)dA)[0] = rA0;
    ((uint4*)dA)[1] = rA1;
    if (TWO) {
      h16* d2 = sA2s[buf] + r * 32 + kh;
      ((uint4*)d2)[0] = rA20;
      ((uint4*)d2)[1] = rA21;
    }
#endif
    union { uint4 u; h16 h[8]; } cv;
    cv.u = rB;
    for (int j = 0; j < 8; ++j) sB1[buf][(bn + j) * 32 + bk] = cv.h[j];
    if (TWO) {
      cv.u = rB2;
      cv.u.x ^= negMask; cv.u.y ^= negMask; cv.u.z ^= negMask; cv.u.w ^= negMask;
      for (int j = 0; j < 8; ++j) sB2s[buf][(bn + j) * 32 + bk] = cv.h[j];
    }
  };

  const int nk = K >> 5;
  stage_issue(0, 0);
  stage_store(0);
#if ASYNC_LDS
  WAIT_ASYNC();
#endif
  __syncthreads();

  for (int kt = 0; kt < nk; ++kt) {
    const int cur = kt & 1;
    const bool pre = (kt + 1 < nk);
    if (pre) stage_issue(kt + 1, cur ^ 1);   // next A -> LDS (async), next B -> regs

    // ---- compute tile kt: 4 (or 8) WMMA per wave ----
    v16h a0 = frag_a(sA1[cur], wm * 32,      lane);
    v16h a1 = frag_a(sA1[cur], wm * 32 + 16, lane);
    v16h b0 = frag_b(sB1[cur], wn * 32,      lane);
    v16h b1 = frag_b(sB1[cur], wn * 32 + 16, lane);
    acc[0][0] = WMMA16(a0, b0, acc[0][0]);
    acc[0][1] = WMMA16(a0, b1, acc[0][1]);
    acc[1][0] = WMMA16(a1, b0, acc[1][0]);
    acc[1][1] = WMMA16(a1, b1, acc[1][1]);
    if (TWO) {
      v16h c0 = frag_a(sA2s[cur], wm * 32,      lane);
      v16h c1 = frag_a(sA2s[cur], wm * 32 + 16, lane);
      v16h d0 = frag_b(sB2s[cur], wn * 32,      lane);
      v16h d1 = frag_b(sB2s[cur], wn * 32 + 16, lane);
      acc[0][0] = WMMA16(c0, d0, acc[0][0]);
      acc[0][1] = WMMA16(c0, d1, acc[0][1]);
      acc[1][0] = WMMA16(c1, d0, acc[1][0]);
      acc[1][1] = WMMA16(c1, d1, acc[1][1]);
    }

    if (pre) {
      stage_store(cur ^ 1);     // B transpose-stores for tile kt+1
#if ASYNC_LDS
      WAIT_ASYNC();             // A async copies for tile kt+1 landed
#endif
    }
    __syncthreads();
  }

  // ---- epilogue: C layout lane=col, vgpr v -> row v + (lane<16?0:8) ----
  const int col = lane & 15;
  const int rb  = (lane < 16) ? 0 : 8;
  const long dz = offD * z;
  const long xz = offX * z;
  for (int tm = 0; tm < 2; ++tm) {
    int gm0 = rowBase + wm * 32 + tm * 16 + rb;
    for (int tn = 0; tn < 2; ++tn) {
      int gn = colBase + wn * 32 + tn * 16 + col;
      if (gn >= Nreal) continue;
      v8f a = acc[tm][tn];
      for (int v = 0; v < 8; ++v) {
        int gm = gm0 + v;
        float val = a[v];
        if (epi == 1)      val = gelu_f(val);
        else if (epi == 2) val += (float)((const h16*)aux)[xz + (long)gm * sXm + (long)gn * sXn];
        else if (epi == 3) val += ((const float*)aux)[xz + (long)gm * sXm + (long)gn * sXn];
        D[dz + (long)gm * sDm + (long)gn * sDn] = (h16)val;
      }
    }
  }
}

// ---------------------------------------------------------------------------
// Pack/pad kernels: f32/f16 -> zero-padded f16, batched over z.
// ---------------------------------------------------------------------------
__global__ __launch_bounds__(256) void pack_pad_f32(
    const float* __restrict__ src, long offS, long sr, long sc,
    h16* __restrict__ dst, long offDz, int R, int C, int Rp, int Cp)
{
  long idx = (long)blockIdx.x * 256 + threadIdx.x;
  long tot = (long)Rp * Cp;
  if (idx >= tot) return;
  int r = (int)(idx / Cp), c = (int)(idx % Cp);
  float v = (r < R && c < C) ? src[offS * blockIdx.z + (long)r * sr + (long)c * sc] : 0.f;
  dst[offDz * blockIdx.z + idx] = (h16)v;
}

__global__ __launch_bounds__(256) void pack_pad_f16(
    const h16* __restrict__ src, long sr,
    h16* __restrict__ dst, int R, int C, int Rp, int Cp)
{
  long idx = (long)blockIdx.x * 256 + threadIdx.x;
  long tot = (long)Rp * Cp;
  if (idx >= tot) return;
  int r = (int)(idx / Cp), c = (int)(idx % Cp);
  h16 v = (r < R && c < C) ? src[(long)r * sr + c] : (h16)0.f;
  dst[idx] = v;
}

// Transpose Fourier coeffs: xf[(c,h)][96 m] -> xfT[96 m][192 h][256 c]
// (h rows 180..191 are never written; Legendre-fwd A is zero there, so they
// contribute nothing). Both real and imag in one pass.
__global__ __launch_bounds__(256) void xpose_xf_kernel(
    const h16* __restrict__ xr, const h16* __restrict__ xi,
    h16* __restrict__ tr, h16* __restrict__ ti)
{
  long idx = (long)blockIdx.x * 256 + threadIdx.x;   // = (c*180+h)*96 + m
  if (idx >= (long)256 * 180 * 96) return;
  int m = (int)(idx % 96);
  long t = idx / 96;
  int hh = (int)(t % 180);
  int c  = (int)(t / 180);
  long dst = (long)m * (192 * 256) + (long)hh * 256 + c;
  tr[dst] = xr[idx];
  ti[dst] = xi[idx];
}

// DFT tables. Forward (cols m contiguous):  Fc[w][m]=cos, Fs[w][m]=-sin
// Inverse (rows m): Ci[m][w]=alpha*cos, Si[m][w]=-alpha*sin ; alpha=(m?2:1)/360
__global__ __launch_bounds__(256) void build_tables_kernel(
    h16* __restrict__ Fc, h16* __restrict__ Fs,
    h16* __restrict__ Ci, h16* __restrict__ Si)
{
  int idx = blockIdx.x * 256 + threadIdx.x;
  if (idx >= 384 * 96) return;
  int w = idx / 96, m = idx % 96;
  bool ok = (w < 360) && (m < 91);
  float th = 6.28318530717958647f * (float)((m * w) % 360) / 360.f;
  float c = ok ? cosf(th) : 0.f;
  float s = ok ? sinf(th) : 0.f;
  Fc[(long)w * 96 + m] = (h16)c;
  Fs[(long)w * 96 + m] = (h16)(-s);
  float al = ((m == 0) ? 1.f : 2.f) / 360.f;
  Ci[(long)m * 384 + w] = (h16)(al * c);
  Si[(long)m * 384 + w] = (h16)(-al * s);
}

// encoder front: t[o][p] = gelu(w1[o,0]*x0[p] + w1[o,1]*x1[p])
__global__ __launch_bounds__(256) void enc_front_kernel(
    const float* __restrict__ x, const float* __restrict__ w1, h16* __restrict__ t)
{
  long idx = (long)blockIdx.x * 256 + threadIdx.x;
  if (idx >= (long)256 * 64800) return;
  int o = (int)(idx / 64800), p = (int)(idx % 64800);
  float v = w1[o * 2] * x[p] + w1[o * 2 + 1] * x[64800 + p];
  t[idx] = (h16)gelu_f(v);
}

// decoder mid: t2 = gelu(s1 + dec_w1[:,256]*x0 + dec_w1[:,257]*x1)
__global__ __launch_bounds__(256) void dec_mid_kernel(
    const h16* __restrict__ s1, const float* __restrict__ x,
    const float* __restrict__ dw1, h16* __restrict__ t2)
{
  long idx = (long)blockIdx.x * 256 + threadIdx.x;
  if (idx >= (long)256 * 64800) return;
  int o = (int)(idx / 64800), p = (int)(idx % 64800);
  float v = (float)s1[idx] + dw1[o * 258 + 256] * x[p] + dw1[o * 258 + 257] * x[64800 + p];
  t2[idx] = (h16)gelu_f(v);
}

// decoder out: out[0/1][p] = sum_c dec_w2[0/1][c] * t2[c][p]   (f32 output)
__global__ __launch_bounds__(256) void dec_out_kernel(
    const h16* __restrict__ t2, const float* __restrict__ w2, float* __restrict__ out)
{
  int p = blockIdx.x * 256 + threadIdx.x;
  if (p >= 64800) return;
  float a0 = 0.f, a1 = 0.f;
  for (int c = 0; c < 256; ++c) {
    float v = (float)t2[(long)c * 64800 + p];
    a0 += w2[c] * v;
    a1 += w2[256 + c] * v;
  }
  out[p] = a0;
  out[64800 + p] = a1;
}

// ---------------------------------------------------------------------------
static void launch_gemm(hipStream_t st, bool two,
                        int M, int Nreal, int K, int Z,
                        const h16* A1, long offA, long sAm,
                        const h16* B1, long offB, long sBk,
                        const h16* A2, const h16* B2, unsigned negMask,
                        h16* D, long offD, long sDm, long sDn,
                        int epi = 0, const void* aux = nullptr,
                        long offX = 0, long sXm = 0, long sXn = 0)
{
  dim3 g((unsigned)((Nreal + 63) / 64), (unsigned)(M / 128), (unsigned)Z), b(256, 1, 1);
  if (two)
    gemm_f16<true><<<g, b, 0, st>>>(A1, offA, sAm, B1, offB, sBk,
                                    A2, B2, negMask, D, offD, sDm, sDn,
                                    Nreal, K, epi, aux, offX, sXm, sXn);
  else
    gemm_f16<false><<<g, b, 0, st>>>(A1, offA, sAm, B1, offB, sBk,
                                     A2, B2, negMask, D, offD, sDm, sDn,
                                     Nreal, K, epi, aux, offX, sXm, sXn);
}

extern "C" void kernel_launch(void* const* d_in, const int* in_sizes, int n_in,
                              void* d_out, int out_size, void* d_ws, size_t ws_size,
                              hipStream_t stream) {
  (void)in_sizes; (void)n_in; (void)out_size; (void)ws_size;

  const float* x       = (const float*)d_in[0];
  const float* enc_w1  = (const float*)d_in[1];
  const float* enc_w2  = (const float*)d_in[2];
  const float* pos     = (const float*)d_in[3];
  const float* leg_fwd = (const float*)d_in[4];
  const float* leg_inv = (const float*)d_in[5];
  const float* wspec_r = (const float*)d_in[6];
  const float* wspec_i = (const float*)d_in[7];
  const float* w_inner = (const float*)d_in[8];
  const float* w_mlp1  = (const float*)d_in[9];
  const float* w_mlp2  = (const float*)d_in[10];
  const float* dec_w1  = (const float*)d_in[11];
  const float* dec_w2  = (const float*)d_in[12];
  float* out = (float*)d_out;

  const int P = 64800;                 // pixels = 180*360
  const long RCH = 46080;              // (c,h) rows = 256*180

  // ---- workspace layout (f16, 256B aligned) ----
  size_t off = 0;
  char* base = (char*)d_ws;
  auto alloc = [&](size_t halfs) -> h16* {
    h16* p = (h16*)(base + off);
    off += ((halfs * 2 + 255) / 256) * 256;
    return p;
  };
  h16* ENC2W = alloc(65536);                       // [256][256]
  h16* DEC1W = alloc(65536);                       // [256][256]
  h16* LEGF  = alloc((size_t)91 * 128 * 192);      // [91][128 l][192 h] zero-pad
  h16* LEGI  = alloc((size_t)91 * 128 * 192);
  h16* WSR2  = alloc((size_t)4 * 90 * 256 * 256);  // [i][l][o][c]
  h16* WSI2  = alloc((size_t)4 * 90 * 256 * 256);
  h16* WIN   = alloc((size_t)4 * 256 * 256);
  h16* WM1   = alloc((size_t)4 * 512 * 256);
  h16* WM2   = alloc((size_t)4 * 256 * 512);
  h16* FC    = alloc((size_t)384 * 96 + 128);      // fwd cos  [w][m]
  h16* FS    = alloc((size_t)384 * 96 + 128);      // fwd -sin [w][m]
  h16* CI    = alloc((size_t)96 * 384 + 128);      // inv cos  [m][w]
  h16* SI    = alloc((size_t)96 * 384 + 128);      // inv -sin [m][w]
  h16* HBUF  = alloc((size_t)256 * P + 4096);      // activations [c][64800]
  h16* HMID  = alloc((size_t)256 * P + 4096);
  h16* ISHT  = alloc((size_t)256 * P + 4096);
  h16* TBUF  = alloc((size_t)512 * P + 4096);      // encoder tmp / MLP hidden
  h16* HPAD  = alloc(RCH * 384 + 4096);            // h padded W->384 for DFT
  h16* XFR   = alloc(RCH * 96 + 8192);             // [(c,h)][96 m]
  h16* XFI   = alloc(RCH * 96 + 8192);
  h16* XFTR  = alloc((size_t)96 * 192 * 256 + 4096); // [m][h][c] transposed coeffs
  h16* XFTI  = alloc((size_t)96 * 192 * 256 + 4096);
  h16* CFR   = alloc((size_t)128 * 256 * 96 + 4096); // fwd coeff [l][c][m]
  h16* CFI   = alloc((size_t)128 * 256 * 96 + 4096);
  h16* C2R   = alloc((size_t)96 * 256 * 96 + 4096);  // filtered [m][o][l]
  h16* C2I   = alloc((size_t)96 * 256 * 96 + 4096);

  auto grid1 = [](long tot) { return dim3((unsigned)((tot + 255) / 256), 1, 1); };
  dim3 blk(256, 1, 1);

  // ---- one-time (per launch) weight packs + tables ----
  pack_pad_f32<<<grid1(65536), blk, 0, stream>>>(enc_w2, 0, 256, 1, ENC2W, 0, 256, 256, 256, 256);
  pack_pad_f32<<<grid1(65536), blk, 0, stream>>>(dec_w1, 0, 258, 1, DEC1W, 0, 256, 256, 256, 256);
  {
    dim3 g((unsigned)(((long)128 * 192 + 255) / 256), 1, 91);
    pack_pad_f32<<<g, blk, 0, stream>>>(leg_fwd, (long)90 * 180, 180, 1, LEGF, (long)128 * 192, 90, 180, 128, 192);
    pack_pad_f32<<<g, blk, 0, stream>>>(leg_inv, (long)90 * 180, 180, 1, LEGI, (long)128 * 192, 90, 180, 128, 192);
  }
  for (int i = 0; i < 4; ++i) {
    // w_spec[i][o][c][l] -> [l][o][c]  (batched over z=l: src offset 1/el)
    dim3 g(256, 1, 90);
    pack_pad_f32<<<g, blk, 0, stream>>>(wspec_r + (long)i * 5898240, 1, 23040, 90,
                                        WSR2 + (long)i * 90 * 65536, 65536, 256, 256, 256, 256);
    pack_pad_f32<<<g, blk, 0, stream>>>(wspec_i + (long)i * 5898240, 1, 23040, 90,
                                        WSI2 + (long)i * 90 * 65536, 65536, 256, 256, 256, 256);
  }
  pack_pad_f32<<<grid1((long)4 * 65536), blk, 0, stream>>>(w_inner, 0, 256, 1, WIN, 0, 1024, 256, 1024, 256);
  pack_pad_f32<<<grid1((long)4 * 131072), blk, 0, stream>>>(w_mlp1, 0, 256, 1, WM1, 0, 2048, 256, 2048, 256);
  pack_pad_f32<<<grid1((long)4 * 131072), blk, 0, stream>>>(w_mlp2, 0, 512, 1, WM2, 0, 1024, 512, 1024, 512);
  build_tables_kernel<<<grid1(384 * 96), blk, 0, stream>>>(FC, FS, CI, SI);

  // ---- encoder: t = gelu(enc_w1 @ x); h = enc_w2 @ t + pos ----
  enc_front_kernel<<<grid1((long)256 * P), blk, 0, stream>>>(x, enc_w1, TBUF);
  launch_gemm(stream, false, 256, P, 256, 1,
              ENC2W, 0, 256, TBUF, 0, P, nullptr, nullptr, 0,
              HBUF, 0, P, 1, /*epi=*/3, pos, 0, P, 1);

  // ---- FNO layers ----
  for (int i = 0; i < 4; ++i) {
    // pad h [c][180][360] -> HPAD [(c,h)][384]
    pack_pad_f16<<<grid1(RCH * 384), blk, 0, stream>>>(HBUF, 360, HPAD, (int)RCH, 360, (int)RCH, 384);

    // DFT forward: XFR/XFI[(c,h)][m] = HPAD @ {FC,FS}
    launch_gemm(stream, false, (int)RCH, 96, 384, 1,
                HPAD, 0, 384, FC, 0, 96, nullptr, nullptr, 0, XFR, 0, 96, 1);
    launch_gemm(stream, false, (int)RCH, 96, 384, 1,
                HPAD, 0, 384, FS, 0, 96, nullptr, nullptr, 0, XFI, 0, 96, 1);

    // transpose to [m][h][c] so Legendre-fwd B is contiguous
    xpose_xf_kernel<<<grid1((long)256 * 180 * 96), blk, 0, stream>>>(XFR, XFI, XFTR, XFTI);

    // Legendre forward (batched over m=91): CF[l][c][m] = LEGF[m] @ XFT[m]
    launch_gemm(stream, false, 128, 256, 192, 91,
                LEGF, (long)128 * 192, 192,
                XFTR, (long)192 * 256, 256, nullptr, nullptr, 0,
                CFR, 1, (long)256 * 96, 96);
    launch_gemm(stream, false, 128, 256, 192, 91,
                LEGF, (long)128 * 192, 192,
                XFTI, (long)192 * 256, 256, nullptr, nullptr, 0,
                CFI, 1, (long)256 * 96, 96);

    // Spectral filter (batched over l=90, complex), weights [l][o][c]:
    // C2R = W_r@CF_r - W_i@CF_i ; C2I = W_r@CF_i + W_i@CF_r ; out [m][o][l]
    launch_gemm(stream, true, 256, 96, 256, 90,
                WSR2 + (long)i * 90 * 65536, 65536, 256,
                CFR, (long)256 * 96, 96,
                WSI2 + (long)i * 90 * 65536, CFI, 0x80008000u,
                C2R, 1, 96, (long)256 * 96);
    launch_gemm(stream, true, 256, 96, 256, 90,
                WSR2 + (long)i * 90 * 65536, 65536, 256,
                CFI, (long)256 * 96, 96,
                WSI2 + (long)i * 90 * 65536, CFR, 0u,
                C2I, 1, 96, (long)256 * 96);

    // Legendre inverse (batched over m=91): XF[.,h,m] = C2[m] @ LEGI[m]
    launch_gemm(stream, false, 256, 180, 96, 91,
                C2R, (long)256 * 96, 96,
                LEGI, (long)128 * 192, 192, nullptr, nullptr, 0,
                XFR, 1, (long)180 * 96, 96);
    launch_gemm(stream, false, 256, 180, 96, 91,
                C2I, (long)256 * 96, 96,
                LEGI, (long)128 * 192, 192, nullptr, nullptr, 0,
                XFI, 1, (long)180 * 96, 96);

    // DFT inverse (two-term): ISHT = XFR@CI + XFI@SI
    launch_gemm(stream, true, (int)RCH, 360, 96, 1,
                XFR, 0, 96, CI, 0, 384,
                XFI, SI, 0u,
                ISHT, 0, 360, 1);

    // inner skip: HMID = W_inner @ h + ISHT
    launch_gemm(stream, false, 256, P, 256, 1,
                WIN + (long)i * 65536, 0, 256, HBUF, 0, P, nullptr, nullptr, 0,
                HMID, 0, P, 1, /*epi=*/2, ISHT, 0, P, 1);

    // MLP: TBUF = gelu(W1 @ HMID); h = W2 @ TBUF + h (outer residual)
    launch_gemm(stream, false, 512, P, 256, 1,
                WM1 + (long)i * 131072, 0, 256, HMID, 0, P, nullptr, nullptr, 0,
                TBUF, 0, P, 1, /*epi=*/1);
    launch_gemm(stream, false, 256, P, 512, 1,
                WM2 + (long)i * 131072, 0, 512, TBUF, 0, P, nullptr, nullptr, 0,
                HBUF, 0, P, 1, /*epi=*/2, HBUF, 0, P, 1);
  }

  // ---- decoder: concat [h; x] -> dec_w1 -> gelu -> dec_w2 ----
  launch_gemm(stream, false, 256, P, 256, 1,
              DEC1W, 0, 256, HBUF, 0, P, nullptr, nullptr, 0,
              HMID, 0, P, 1);
  dec_mid_kernel<<<grid1((long)256 * P), blk, 0, stream>>>(HMID, x, dec_w1, TBUF);
  dec_out_kernel<<<grid1(P), blk, 0, stream>>>(TBUF, dec_w2, out);
}
